// PointNet2FeaExtractor_979252544038
// MI455X (gfx1250) — matrix operand order repaired and checked by reading, not compiled, not run on hardware
//
#include <hip/hip_runtime.h>
#include <math.h>

typedef __attribute__((ext_vector_type(16))) _Float16 v16h;
typedef __attribute__((ext_vector_type(8)))  float    v8f;

#define BB  2
#define NN  8192
#define CF0 10
#define S1  4096
#define S2  1024
#define NSAMP 16

// ---------------- workspace arena (float units) ----------------
static const size_t OFF_L1X   = 0;                      // 24576 -> pad 32768
static const size_t OFF_L1F   = OFF_L1X   + 32768;      // 262144
static const size_t OFF_L2X   = OFF_L1F   + 262144;     // 6144 -> 8192
static const size_t OFF_L2F   = OFF_L2X   + 8192;       // 131072
static const size_t OFF_L1N   = OFF_L2F   + 131072;     // 262144
static const size_t OFF_SUMAX = OFF_L1N   + 262144;     // 262144
static const size_t OFF_STATS = OFF_SUMAX + 262144;     // 1024
static const size_t OFF_IFPS1 = OFF_STATS + 1024;       // 8192 (int)
static const size_t OFF_GIDX1 = OFF_IFPS1 + 8192;       // 131072 (int)
static const size_t OFF_IFPS2 = OFF_GIDX1 + 131072;     // 4096 (int)
static const size_t OFF_GIDX2 = OFF_IFPS2 + 4096;       // 32768 (int)
static const size_t OFF_KNNSU = OFF_GIDX2 + 32768;      // 65536 (int)
static const size_t OFF_KNNFPI= OFF_KNNSU + 65536;      // 49152 (int)
static const size_t OFF_KNNFPD= OFF_KNNFPI+ 49152;      // 49152
static const size_t OFF_TMPA  = OFF_KNNFPD+ 49152;      // cap 4401152 (need 4390912)
static const size_t OFF_TMPB  = OFF_TMPA  + 4401152;    // 4194304
// total ~9.9M floats ~ 40 MB

static inline int cdiv(long long a, int b){ return (int)((a + b - 1) / b); }

// ---------------- Farthest Point Sampling ----------------
__global__ __launch_bounds__(1024) void fps_kernel(const float* __restrict__ xyz,
                                                   int N, int S, int* __restrict__ idx_out){
  int b   = blockIdx.x;
  int tid = threadIdx.x;
  int nt  = blockDim.x;
  int PT  = (N + nt - 1) / nt;           // <= 8
  const float* p = xyz + (long long)b * N * 3;

  float px[8], py[8], pz[8], dl[8];
  for (int j = 0; j < PT; ++j){
    int pi = tid * PT + j;
    if (pi < N){
      px[j] = p[pi*3+0]; py[j] = p[pi*3+1]; pz[j] = p[pi*3+2];
      dl[j] = 1e10f;
    }
  }
  __shared__ float wv[32];
  __shared__ int   wi[32];
  __shared__ int   sFar;
  __shared__ float scx, scy, scz;
  if (tid == 0) sFar = 0;
  __syncthreads();

  int lane = tid & 31, wid = tid >> 5, nw = nt >> 5;
  for (int s = 0; s < S; ++s){
    int far = sFar;
    if (tid == 0){
      idx_out[(long long)b*S + s] = far;
      scx = p[far*3+0]; scy = p[far*3+1]; scz = p[far*3+2];
    }
    __syncthreads();
    float cx = scx, cy = scy, cz = scz;
    float bv = -1.0f; int bix = 0x7fffffff;
    for (int j = 0; j < PT; ++j){
      int pi = tid * PT + j;
      if (pi < N){
        float dx = px[j]-cx, dy = py[j]-cy, dz = pz[j]-cz;
        float d  = dx*dx + dy*dy + dz*dz;
        if (d < dl[j]) dl[j] = d;
        if (dl[j] > bv){ bv = dl[j]; bix = pi; }  // strict > keeps first max
      }
    }
    for (int off = 16; off > 0; off >>= 1){
      float ov = __shfl_down(bv, off, 32);
      int   oi = __shfl_down(bix, off, 32);
      if (ov > bv || (ov == bv && oi < bix)){ bv = ov; bix = oi; }
    }
    if (lane == 0){ wv[wid] = bv; wi[wid] = bix; }
    __syncthreads();
    if (wid == 0){
      float v2 = (lane < nw) ? wv[lane] : -2.0f;
      int   i2 = (lane < nw) ? wi[lane] : 0x7fffffff;
      for (int off = 16; off > 0; off >>= 1){
        float ov = __shfl_down(v2, off, 32);
        int   oi = __shfl_down(i2, off, 32);
        if (ov > v2 || (ov == v2 && oi < i2)){ v2 = ov; i2 = oi; }
      }
      if (lane == 0) sFar = i2;
    }
    __syncthreads();
  }
}

// ---------------- gather points by index ----------------
__global__ void gather_pts_kernel(const float* __restrict__ src, const int* __restrict__ idx,
                                  int Nsrc, int S, int total, float* __restrict__ out){
  int i = blockIdx.x * blockDim.x + threadIdx.x;
  if (i >= total) return;
  int b = i / S;
  long long sp = (long long)b * Nsrc + idx[i];
  out[(long long)i*3+0] = src[sp*3+0];
  out[(long long)i*3+1] = src[sp*3+1];
  out[(long long)i*3+2] = src[sp*3+2];
}

// ---------------- ball query ----------------
__global__ void ball_query_kernel(const float* __restrict__ src, const float* __restrict__ cen,
                                  int Nsrc, int S, float r2, int* __restrict__ out){
  int b = blockIdx.y;
  int s = blockIdx.x * blockDim.x + threadIdx.x;
  __shared__ float tx[256], ty[256], tz[256];
  float cx = 0.f, cy = 0.f, cz = 0.f;
  bool active = (s < S);
  if (active){
    long long ci = ((long long)b * S + s) * 3;
    cx = cen[ci]; cy = cen[ci+1]; cz = cen[ci+2];
  }
  int nbr[NSAMP]; int cnt = 0;
  float bestd = 1e30f; int besti = 0;
  for (int t0 = 0; t0 < Nsrc; t0 += 256){
    int li = t0 + threadIdx.x;
    if (li < Nsrc){
      long long sp = ((long long)b * Nsrc + li) * 3;
      tx[threadIdx.x] = src[sp]; ty[threadIdx.x] = src[sp+1]; tz[threadIdx.x] = src[sp+2];
    }
    __syncthreads();
    int lim = Nsrc - t0; if (lim > 256) lim = 256;
    if (active){
      for (int j = 0; j < lim; ++j){
        float dx = tx[j]-cx, dy = ty[j]-cy, dz = tz[j]-cz;
        float d  = dx*dx + dy*dy + dz*dz;
        if (d < bestd){ bestd = d; besti = t0 + j; }
        if (d <= r2 && cnt < NSAMP){ nbr[cnt++] = t0 + j; }
      }
    }
    __syncthreads();
  }
  if (active){
    int fill = (cnt > 0) ? nbr[0] : besti;
    long long base = ((long long)b * S + s) * NSAMP;
    for (int j = 0; j < NSAMP; ++j) out[base + j] = (j < cnt) ? nbr[j] : fill;
  }
}

// ---------------- k-NN (stable: ties keep lower index) ----------------
template<int KN>
__global__ void knn_kernel(const float* __restrict__ p1, const float* __restrict__ p2,
                           int N1, int N2, int* __restrict__ idx_out, float* __restrict__ dist_out){
  int b = blockIdx.y;
  int n = blockIdx.x * blockDim.x + threadIdx.x;
  __shared__ float tx[256], ty[256], tz[256];
  float cx = 0.f, cy = 0.f, cz = 0.f;
  bool active = (n < N1);
  if (active){
    long long ci = ((long long)b * N1 + n) * 3;
    cx = p1[ci]; cy = p1[ci+1]; cz = p1[ci+2];
  }
  float bd[KN]; int bi[KN];
  for (int j = 0; j < KN; ++j){ bd[j] = 1e30f; bi[j] = 0; }
  for (int t0 = 0; t0 < N2; t0 += 256){
    int li = t0 + threadIdx.x;
    if (li < N2){
      long long sp = ((long long)b * N2 + li) * 3;
      tx[threadIdx.x] = p2[sp]; ty[threadIdx.x] = p2[sp+1]; tz[threadIdx.x] = p2[sp+2];
    }
    __syncthreads();
    int lim = N2 - t0; if (lim > 256) lim = 256;
    if (active){
      for (int j = 0; j < lim; ++j){
        float dx = tx[j]-cx, dy = ty[j]-cy, dz = tz[j]-cz;
        float d  = dx*dx + dy*dy + dz*dz;
        if (d < bd[KN-1]){
          int pos = KN - 1;
          while (pos > 0 && d < bd[pos-1]){ bd[pos] = bd[pos-1]; bi[pos] = bi[pos-1]; --pos; }
          bd[pos] = d; bi[pos] = t0 + j;
        }
      }
    }
    __syncthreads();
  }
  if (active){
    long long base = ((long long)b * N1 + n) * KN;
    for (int j = 0; j < KN; ++j){
      idx_out[base + j] = bi[j];
      if (dist_out) dist_out[base + j] = bd[j];
    }
  }
}

// ---------------- grouping: h = [xyz - center, feat] ----------------
__global__ void group_sa_kernel(const float* __restrict__ src, const float* __restrict__ feat,
                                const int* __restrict__ gidx, const float* __restrict__ cen,
                                int Nsrc, int S, int rows, int K, int Cf, float* __restrict__ H){
  int row = blockIdx.x * blockDim.x + threadIdx.x;
  if (row >= rows) return;
  int bs = row / K;
  int b  = bs / S;
  int g  = gidx[row];
  long long sp = (long long)b * Nsrc + g;
  int cin = 3 + Cf;
  float* h = H + (long long)row * cin;
  h[0] = src[sp*3+0] - cen[(long long)bs*3+0];
  h[1] = src[sp*3+1] - cen[(long long)bs*3+1];
  h[2] = src[sp*3+2] - cen[(long long)bs*3+2];
  const float* f = feat + sp * Cf;
  for (int c = 0; c < Cf; ++c) h[3+c] = f[c];
}

// ---------------- set_upconv grouping: h = [feat2, pos2 - pos1] ----------------
__global__ void group_su_kernel(const float* __restrict__ feat2, const float* __restrict__ pos2,
                                const float* __restrict__ pos1, const int* __restrict__ idx,
                                int N1, int N2, int Cf, int K, int rows, float* __restrict__ H){
  int row = blockIdx.x * blockDim.x + threadIdx.x;
  if (row >= rows) return;
  int bn = row / K;
  int b  = bn / N1;
  int g  = idx[row];
  long long p2 = (long long)b * N2 + g;
  int cin = Cf + 3;
  float* h = H + (long long)row * cin;
  const float* f = feat2 + p2 * Cf;
  for (int c = 0; c < Cf; ++c) h[c] = f[c];
  h[Cf+0] = pos2[p2*3+0] - pos1[(long long)bn*3+0];
  h[Cf+1] = pos2[p2*3+1] - pos1[(long long)bn*3+1];
  h[Cf+2] = pos2[p2*3+2] - pos1[(long long)bn*3+2];
}

// ---------------- WMMA GEMM: Y(M,COUT) = X(M,CIN) * W(COUT,CIN)^T ----------------
// Compile-time shapes; K-padding uses clamped addresses + value select (no
// exec-masked loads, lanes stay converged -> batched loads, EXEC all-1s for wmma).
template<int CIN, int COUT>
__global__ __launch_bounds__(128) void wmma_gemm_kernel(const float* __restrict__ X,
                                                        const float* __restrict__ W,
                                                        float* __restrict__ Y){
  constexpr int NK = (CIN + 31) / 32;
  int lane    = threadIdx.x & 31;
  int wave    = threadIdx.x >> 5;
  int rowBase = blockIdx.x * 64 + wave * 16;
  int hi      = (lane >= 16);
  int col     = lane & 15;
  const float* xr = X + (long long)(rowBase + col) * CIN;

  // A fragments: loaded once, reused for every COUT tile.
  v16h a[NK];
#pragma unroll
  for (int kk = 0; kk < NK; ++kk){
    int ka = kk * 32 + (hi ? 8 : 0);
#pragma unroll
    for (int j = 0; j < 8; ++j){
      int k  = ka + j;
      int ks = (k < CIN) ? k : 0;               // always-valid address
      float v = xr[ks];
      a[kk][j]   = (_Float16)((k < CIN) ? v : 0.0f);
    }
#pragma unroll
    for (int j = 0; j < 8; ++j){
      int k  = ka + 16 + j;
      int ks = (k < CIN) ? k : 0;
      float v = xr[ks];
      a[kk][8+j] = (_Float16)((k < CIN) ? v : 0.0f);
    }
  }

#pragma unroll
  for (int n0 = 0; n0 < COUT; n0 += 16){
    const float* wr = W + (long long)(n0 + col) * CIN;
    v8f acc = {};
#pragma unroll
    for (int kk = 0; kk < NK; ++kk){
      int kb = kk * 32 + (hi ? 8 : 0);
      v16h bb;
#pragma unroll
      for (int j = 0; j < 16; ++j){
        int k  = kb + (j < 8 ? j : j + 8);
        int ks = (k < CIN) ? k : 0;
        float v = wr[ks];
        bb[j] = (_Float16)((k < CIN) ? v : 0.0f);
      }
      acc = __builtin_amdgcn_wmma_f32_16x16x32_f16(false, a[kk], false, bb,
                                                   (short)0, acc, false, false);
    }
#pragma unroll
    for (int v = 0; v < 8; ++v){
      int rr = rowBase + v + (hi ? 8 : 0);
      Y[(long long)rr * COUT + n0 + col] = acc[v];
    }
  }
}

// ---------------- BatchNorm (global stats) + ReLU ----------------
__global__ __launch_bounds__(256) void bn_stats_kernel(const float* __restrict__ Y,
                                                       long long total, int C,
                                                       float* __restrict__ sums,
                                                       float* __restrict__ sumsq){
  __shared__ float ls[64], lq[64];
  if (threadIdx.x < 64u){ ls[threadIdx.x] = 0.f; lq[threadIdx.x] = 0.f; }
  __syncthreads();
  long long i0 = (long long)blockIdx.x * blockDim.x + threadIdx.x;
  int c = (int)(i0 & (long long)(C - 1));       // C is a power of two (16/32/64)
  float s = 0.f, q = 0.f;
  long long stride = (long long)gridDim.x * blockDim.x;   // multiple of C
  for (long long i = i0; i < total; i += stride){
    float v = Y[i];
    s += v; q += v * v;
  }
  atomicAdd(&ls[c], s);
  atomicAdd(&lq[c], q);
  __syncthreads();
  if (threadIdx.x < (unsigned)C){
    atomicAdd(&sums[threadIdx.x],  ls[threadIdx.x]);
    atomicAdd(&sumsq[threadIdx.x], lq[threadIdx.x]);
  }
}

__global__ void bn_finalize_kernel(const float* __restrict__ sums, const float* __restrict__ sumsq,
                                   const float* __restrict__ g, const float* __restrict__ b,
                                   float invM, int C, float* __restrict__ scale, float* __restrict__ shift){
  int c = threadIdx.x;
  if (c < C){
    float mean = sums[c]  * invM;
    float var  = sumsq[c] * invM - mean * mean;
    float sc   = g[c] * rsqrtf(var + 1e-5f);
    scale[c] = sc;
    shift[c] = b[c] - mean * sc;
  }
}

__global__ void bn_apply_relu_kernel(float* __restrict__ Y, long long total, int C,
                                     const float* __restrict__ scale, const float* __restrict__ shift){
  long long i = (long long)blockIdx.x * blockDim.x + threadIdx.x;
  if (i >= total) return;
  int c = (int)(i & (long long)(C - 1));   // C is a power of two here
  Y[i] = fmaxf(Y[i] * scale[c] + shift[c], 0.0f);
}

// ---------------- max over the K (sample) axis ----------------
__global__ void maxpool_kernel(const float* __restrict__ Y, int Stot, int K, int C,
                               float* __restrict__ out){
  int i = blockIdx.x * blockDim.x + threadIdx.x;
  if (i >= Stot * C) return;
  int s = i / C, c = i % C;
  float m = -1e30f;
  for (int k = 0; k < K; ++k){
    float v = Y[((long long)s * K + k) * C + c];
    m = fmaxf(m, v);
  }
  out[i] = m;
}

// ---------------- channel-wise concat ----------------
__global__ void concat2_kernel(const float* __restrict__ A, const float* __restrict__ B2,
                               int Ca, int Cb, long long Mrows, float* __restrict__ Out){
  long long i = (long long)blockIdx.x * blockDim.x + threadIdx.x;
  long long total = Mrows * (Ca + Cb);
  if (i >= total) return;
  long long r = i / (Ca + Cb);
  int       c = (int)(i % (Ca + Cb));
  Out[i] = (c < Ca) ? A[r * Ca + c] : B2[r * Cb + (c - Ca)];
}

// ---------------- 3-NN inverse-distance interpolation ----------------
__global__ void fp_interp_kernel(const float* __restrict__ f2, const int* __restrict__ idx,
                                 const float* __restrict__ dist, int N1, int N2, int C,
                                 long long total, float* __restrict__ out){
  long long i = (long long)blockIdx.x * blockDim.x + threadIdx.x;
  if (i >= total) return;
  long long bn = i / C;
  int c  = (int)(i % C);
  int b  = (int)(bn / N1);
  float w0 = 1.0f / fmaxf(dist[bn*3+0], 1e-10f);
  float w1 = 1.0f / fmaxf(dist[bn*3+1], 1e-10f);
  float w2 = 1.0f / fmaxf(dist[bn*3+2], 1e-10f);
  float inv = 1.0f / (w0 + w1 + w2);
  long long base = (long long)b * N2;
  float acc = w0 * inv * f2[(base + idx[bn*3+0]) * C + c]
            + w1 * inv * f2[(base + idx[bn*3+1]) * C + c]
            + w2 * inv * f2[(base + idx[bn*3+2]) * C + c];
  out[i] = acc;
}

// ---------------- final concat [pts, sf + bias] ----------------
__global__ void final_out_kernel(const float* __restrict__ pts, const float* __restrict__ sf,
                                 const float* __restrict__ bias, int Npts, float* __restrict__ out){
  long long i = (long long)blockIdx.x * blockDim.x + threadIdx.x;
  long long total = (long long)Npts * 19;
  if (i >= total) return;
  long long p = i / 19;
  int       c = (int)(i % 19);
  out[i] = (c < 3) ? pts[p*3 + c] : sf[p*16 + (c-3)] + bias[c-3];
}

// ================= host orchestration =================
struct Weights {
  const float *sa1_w0,*sa1_g0,*sa1_b0,*sa1_w1,*sa1_g1,*sa1_b1;
  const float *sa2_w0,*sa2_g0,*sa2_b0,*sa2_w1,*sa2_g1,*sa2_b1;
  const float *su1_w1,*su1_g1,*su1_b1,*su1_w2,*su1_g2,*su1_b2;
  const float *fp_w,*fp_g,*fp_b,*bn1_g,*bn1_b,*conv2_w,*conv2_b;
};

template<int CIN, int COUT>
static void run_gemm(const float* X, const float* W, float* Y, int M, hipStream_t st){
  // M is a multiple of 64 at every call site.
  wmma_gemm_kernel<CIN, COUT><<<M / 64, 128, 0, st>>>(X, W, Y);
}

static void run_bn(float* Y, long long M, int C, const float* g, const float* b,
                   float* stats, hipStream_t st){
  hipMemsetAsync(stats, 0, 2 * 64 * sizeof(float), st);
  long long total = M * C;
  bn_stats_kernel<<<256, 256, 0, st>>>(Y, total, C, stats, stats + 64);
  bn_finalize_kernel<<<1, 64, 0, st>>>(stats, stats + 64, g, b, 1.0f / (float)M, C,
                                       stats + 128, stats + 192);
  bn_apply_relu_kernel<<<cdiv(total, 256), 256, 0, st>>>(Y, total, C, stats + 128, stats + 192);
}

static void run_pipeline(const float* pts, const float* fea, const Weights& w,
                         float* ws, float* out, hipStream_t st){
  float* l1x   = ws + OFF_L1X;
  float* l1f   = ws + OFF_L1F;
  float* l2x   = ws + OFF_L2X;
  float* l2f   = ws + OFF_L2F;
  float* l1n   = ws + OFF_L1N;
  float* sumax = ws + OFF_SUMAX;
  float* stats = ws + OFF_STATS;
  int*   ifps1 = (int*)(ws + OFF_IFPS1);
  int*   gidx1 = (int*)(ws + OFF_GIDX1);
  int*   ifps2 = (int*)(ws + OFF_IFPS2);
  int*   gidx2 = (int*)(ws + OFF_GIDX2);
  int*   knnsu = (int*)(ws + OFF_KNNSU);
  int*   knnfpi= (int*)(ws + OFF_KNNFPI);
  float* knnfpd= ws + OFF_KNNFPD;
  float* tA    = ws + OFF_TMPA;
  float* tB    = ws + OFF_TMPB;

  // ---- set_abstraction level 1 ----
  fps_kernel<<<BB, 1024, 0, st>>>(pts, NN, S1, ifps1);
  gather_pts_kernel<<<cdiv((long long)BB*S1, 256), 256, 0, st>>>(pts, ifps1, NN, S1, BB*S1, l1x);
  ball_query_kernel<<<dim3(S1/256, BB), 256, 0, st>>>(pts, l1x, NN, S1, 0.1f*0.1f, gidx1);
  {
    int rows = BB * S1 * NSAMP;                                  // 131072
    group_sa_kernel<<<cdiv(rows, 256), 256, 0, st>>>(pts, fea, gidx1, l1x, NN, S1, rows, NSAMP, CF0, tA);
    run_gemm<13, 16>(tA, w.sa1_w0, tB, rows, st);
    run_bn(tB, rows, 16, w.sa1_g0, w.sa1_b0, stats, st);
    run_gemm<16, 32>(tB, w.sa1_w1, tA, rows, st);
    run_bn(tA, rows, 32, w.sa1_g1, w.sa1_b1, stats, st);
    maxpool_kernel<<<cdiv((long long)BB*S1*32, 256), 256, 0, st>>>(tA, BB*S1, NSAMP, 32, l1f);
  }
  // ---- set_abstraction level 2 ----
  fps_kernel<<<BB, 1024, 0, st>>>(l1x, S1, S2, ifps2);
  gather_pts_kernel<<<cdiv((long long)BB*S2, 256), 256, 0, st>>>(l1x, ifps2, S1, S2, BB*S2, l2x);
  ball_query_kernel<<<dim3(S2/256, BB), 256, 0, st>>>(l1x, l2x, S1, S2, 0.2f*0.2f, gidx2);
  {
    int rows = BB * S2 * NSAMP;                                  // 32768
    group_sa_kernel<<<cdiv(rows, 256), 256, 0, st>>>(l1x, l1f, gidx2, l2x, S1, S2, rows, NSAMP, 32, tB);
    run_gemm<35, 32>(tB, w.sa2_w0, tA, rows, st);
    run_bn(tA, rows, 32, w.sa2_g0, w.sa2_b0, stats, st);
    run_gemm<32, 64>(tA, w.sa2_w1, tB, rows, st);
    run_bn(tB, rows, 64, w.sa2_g1, w.sa2_b1, stats, st);
    maxpool_kernel<<<cdiv((long long)BB*S2*64, 256), 256, 0, st>>>(tB, BB*S2, NSAMP, 64, l2f);
  }
  // ---- set_upconv ----
  knn_kernel<8><<<dim3(S1/256, BB), 256, 0, st>>>(l1x, l2x, S1, S2, knnsu, (float*)nullptr);
  {
    int rows = BB * S1 * 8;                                      // 65536
    group_su_kernel<<<cdiv(rows, 256), 256, 0, st>>>(l2f, l2x, l1x, knnsu, S1, S2, 64, 8, rows, tA);
    run_gemm<67, 32>(tA, w.su1_w1, tB, rows, st);
    run_bn(tB, rows, 32, w.su1_g1, w.su1_b1, stats, st);
    maxpool_kernel<<<cdiv((long long)BB*S1*32, 256), 256, 0, st>>>(tB, BB*S1, 8, 32, sumax);
    concat2_kernel<<<cdiv((long long)BB*S1*64, 256), 256, 0, st>>>(sumax, l1f, 32, 32, BB*S1, tA);
    run_gemm<64, 32>(tA, w.su1_w2, l1n, BB*S1, st);
    run_bn(l1n, (long long)BB*S1, 32, w.su1_g2, w.su1_b2, stats, st);
  }
  // ---- feature propagation + head ----
  knn_kernel<3><<<dim3(NN/256, BB), 256, 0, st>>>(pts, l1x, NN, S1, knnfpi, knnfpd);
  {
    long long ti = (long long)BB * NN * 32;
    fp_interp_kernel<<<cdiv(ti, 256), 256, 0, st>>>(l1n, knnfpi, knnfpd, NN, S1, 32, ti, tB);
    concat2_kernel<<<cdiv((long long)BB*NN*42, 256), 256, 0, st>>>(tB, fea, 32, CF0, BB*NN, tA);
    run_gemm<42, 32>(tA, w.fp_w, tB, BB*NN, st);
    run_bn(tB, (long long)BB*NN, 32, w.fp_g,  w.fp_b,  stats, st);   // feature_prop bn_relu
    run_bn(tB, (long long)BB*NN, 32, w.bn1_g, w.bn1_b, stats, st);   // outer bn1 relu
    run_gemm<32, 16>(tB, w.conv2_w, tA, BB*NN, st);
    final_out_kernel<<<cdiv((long long)BB*NN*19, 256), 256, 0, st>>>(pts, tA, w.conv2_b, BB*NN, out);
  }
}

extern "C" void kernel_launch(void* const* d_in, const int* in_sizes, int n_in,
                              void* d_out, int out_size, void* d_ws, size_t ws_size,
                              hipStream_t stream){
  (void)in_sizes; (void)n_in; (void)out_size; (void)ws_size;
  const float* pts1 = (const float*)d_in[0];
  const float* fea1 = (const float*)d_in[1];
  const float* pts2 = (const float*)d_in[2];
  const float* fea2 = (const float*)d_in[3];
  Weights w;
  w.sa1_w0 = (const float*)d_in[4];  w.sa1_g0 = (const float*)d_in[5];  w.sa1_b0 = (const float*)d_in[6];
  w.sa1_w1 = (const float*)d_in[7];  w.sa1_g1 = (const float*)d_in[8];  w.sa1_b1 = (const float*)d_in[9];
  w.sa2_w0 = (const float*)d_in[10]; w.sa2_g0 = (const float*)d_in[11]; w.sa2_b0 = (const float*)d_in[12];
  w.sa2_w1 = (const float*)d_in[13]; w.sa2_g1 = (const float*)d_in[14]; w.sa2_b1 = (const float*)d_in[15];
  w.su1_w1 = (const float*)d_in[16]; w.su1_g1 = (const float*)d_in[17]; w.su1_b1 = (const float*)d_in[18];
  w.su1_w2 = (const float*)d_in[19]; w.su1_g2 = (const float*)d_in[20]; w.su1_b2 = (const float*)d_in[21];
  w.fp_w   = (const float*)d_in[22]; w.fp_g   = (const float*)d_in[23]; w.fp_b   = (const float*)d_in[24];
  w.bn1_g  = (const float*)d_in[25]; w.bn1_b  = (const float*)d_in[26];
  w.conv2_w= (const float*)d_in[27]; w.conv2_b= (const float*)d_in[28];

  float* ws  = (float*)d_ws;
  float* out = (float*)d_out;
  run_pipeline(pts1, fea1, w, ws, out, stream);
  run_pipeline(pts2, fea2, w, ws, out + (size_t)BB * NN * 19, stream);
}
